// PairProposalNetwork_87857851007559
// MI455X (gfx1250) — matrix operand order repaired and checked by reading, not compiled, not run on hardware
//
#include <hip/hip_runtime.h>
#include <math.h>

typedef __attribute__((ext_vector_type(16))) _Float16 v16h;
typedef __attribute__((ext_vector_type(8)))  float    v8f;

#define BB    16
#define NN    300
#define CC    256
#define NCLS  134      // NUM_CLS + 1
#define KREL  100
#define MID   64
#define PIX   90000    // 300*300
#define K2    3136     // 64*49
#define MT2   (PIX/16) // 5625 M-tiles

// ---------------------------------------------------------------------------
// Small utility kernels
// ---------------------------------------------------------------------------
__global__ void f32_to_f16_kernel(const float* __restrict__ x,
                                  _Float16* __restrict__ y, int n) {
    int i = blockIdx.x * blockDim.x + threadIdx.x;
    if (i < n) y[i] = (_Float16)x[i];
}

// Wt[n*K + k] = W[k*N + n]  (f32 -> f16), for WMMA B-fragment contiguity
__global__ void transpose_to_f16_kernel(const float* __restrict__ W,
                                        _Float16* __restrict__ Wt, int K, int N) {
    int i = blockIdx.x * blockDim.x + threadIdx.x;
    if (i < K * N) {
        int n = i / K, k = i - n * K;
        Wt[i] = (_Float16)W[(size_t)k * N + n];
    }
}

// conv1 weights: wp[oc][k] (64x64 f16), taps 0..48 real, 49..63 zero-pad
__global__ void conv1_wprep_kernel(const float* __restrict__ w,
                                   _Float16* __restrict__ wp) {
    int i = blockIdx.x * blockDim.x + threadIdx.x;   // over 64*64
    if (i < MID * 64) {
        int oc = i >> 6, t = i & 63;
        wp[i] = (t < 49) ? (_Float16)w[oc * 49 + t] : (_Float16)0.0f;
    }
}

// conv2 weight reorder: wr[oc][tap*64 + ic] = cw2[oc][ic][tap]  (f32 -> f16)
__global__ void conv2_wreorder_kernel(const float* __restrict__ w,
                                      _Float16* __restrict__ wr) {
    int i = blockIdx.x * blockDim.x + threadIdx.x;  // over 64*3136
    if (i < MID * K2) {
        int oc = i / K2, r = i - oc * K2;
        int t = r >> 6, ic = r & 63;                // r = t*64 + ic
        wr[i] = (_Float16)w[(size_t)oc * K2 + ic * 49 + t];
    }
}

// conv3 weights: wp[row][tap*64+ic], row 0 = cw3[ic][tap], rows 1..15 zero
__global__ void conv3_wprep_kernel(const float* __restrict__ w,
                                   _Float16* __restrict__ wp) {
    int i = blockIdx.x * blockDim.x + threadIdx.x;   // over 16*3136
    if (i < 16 * K2) {
        int row = i / K2, r = i - row * K2;
        int t = r >> 6, ic = r & 63;
        wp[i] = (row == 0) ? (_Float16)w[ic * 49 + t] : (_Float16)0.0f;
    }
}

// ---------------------------------------------------------------------------
// MLP layer: Y = act(X @ W + b); X:(rows,256) f16, Wt:(256,256) f16 [n][k]
// ---------------------------------------------------------------------------
__global__ void mlp_layer_kernel(const _Float16* __restrict__ X,
                                 const _Float16* __restrict__ Wt,
                                 const float* __restrict__ bias,
                                 _Float16* __restrict__ Y,
                                 int rows, int doRelu) {
    int wave = (blockIdx.x * blockDim.x + threadIdx.x) >> 5;
    int lane = threadIdx.x & 31;
    int mTiles = rows >> 4;
    int tileN = wave & 15;
    int tileM = wave >> 4;
    if (tileM >= mTiles) return;

    int mrow = tileM * 16 + (lane & 15);
    int ncol = tileN * 16 + (lane & 15);
    int ahi  = (lane >= 16) ? 8 : 0;
    int bhi  = (lane >= 16) ? 8 : 0;

    v8f c = {};
    for (int k0 = 0; k0 < CC; k0 += 32) {
        v16h a, b;
        const _Float16* xr = X  + (size_t)mrow * CC + k0;
        const _Float16* wr = Wt + (size_t)ncol * CC + k0 + bhi;
        #pragma unroll
        for (int h = 0; h < 16; ++h) {
            int ka = ((h < 8) ? h : 16 + (h - 8)) + ahi;
            a[h] = xr[ka];
            b[h] = wr[(h < 8) ? h : 16 + (h - 8)];
        }
        c = __builtin_amdgcn_wmma_f32_16x16x32_f16(false, a, false, b,
                                                   (short)0, c, false, false);
    }
    float bv = bias[ncol];
    #pragma unroll
    for (int r = 0; r < 8; ++r) {
        int m = tileM * 16 + r + ((lane >= 16) ? 8 : 0);
        float v = c[r] + bv;
        if (doRelu) v = fmaxf(v, 0.0f);
        Y[(size_t)m * CC + ncol] = (_Float16)v;
    }
}

// ---------------------------------------------------------------------------
// Row-wise L2 normalize (one wave per 256-wide row)
// ---------------------------------------------------------------------------
__global__ void l2norm_kernel(const _Float16* __restrict__ X,
                              _Float16* __restrict__ Y, int rows) {
    int wave = (blockIdx.x * blockDim.x + threadIdx.x) >> 5;
    int lane = threadIdx.x & 31;
    if (wave >= rows) return;
    const _Float16* xr = X + (size_t)wave * CC;
    float s = 0.0f;
    #pragma unroll
    for (int i = 0; i < CC / 32; ++i) {
        float v = (float)xr[lane + i * 32];
        s += v * v;
    }
    #pragma unroll
    for (int off = 16; off > 0; off >>= 1) s += __shfl_xor(s, off, 32);
    float inv = 1.0f / fmaxf(sqrtf(s), 1e-12f);
    _Float16* yr = Y + (size_t)wave * CC;
    #pragma unroll
    for (int i = 0; i < CC / 32; ++i)
        yr[lane + i * 32] = (_Float16)((float)xr[lane + i * 32] * inv);
}

// ---------------------------------------------------------------------------
// Batched einsum: out[b] = sub[b] (300x256) @ obj[b]^T -> (300,300) f16 image
// grid (19,19,B), block = 32 (one wave per 16x16 tile)
// ---------------------------------------------------------------------------
__global__ void einsum_kernel(const _Float16* __restrict__ sub,
                              const _Float16* __restrict__ obj,
                              _Float16* __restrict__ out) {
    int b = blockIdx.z;
    int tileM = blockIdx.x, tileN = blockIdx.y;
    int lane = threadIdx.x & 31;
    const _Float16* S = sub + (size_t)b * NN * CC;
    const _Float16* O = obj + (size_t)b * NN * CC;

    int mrow = tileM * 16 + (lane & 15);
    int nrow = tileN * 16 + (lane & 15);
    bool mok = mrow < NN, nok = nrow < NN;
    int ahi = (lane >= 16) ? 8 : 0;
    int bhi = (lane >= 16) ? 8 : 0;

    v8f c = {};
    for (int k0 = 0; k0 < CC; k0 += 32) {
        v16h a = {}, bfrag = {};
        if (mok) {
            const _Float16* xr = S + (size_t)mrow * CC + k0;
            #pragma unroll
            for (int h = 0; h < 16; ++h)
                a[h] = xr[((h < 8) ? h : 16 + (h - 8)) + ahi];
        }
        if (nok) {
            const _Float16* yr = O + (size_t)nrow * CC + k0 + bhi;
            #pragma unroll
            for (int h = 0; h < 16; ++h) bfrag[h] = yr[(h < 8) ? h : 16 + (h - 8)];
        }
        c = __builtin_amdgcn_wmma_f32_16x16x32_f16(false, a, false, bfrag,
                                                   (short)0, c, false, false);
    }
    if (nrow < NN) {
        #pragma unroll
        for (int r = 0; r < 8; ++r) {
            int m = tileM * 16 + r + ((lane >= 16) ? 8 : 0);
            if (m < NN)
                out[((size_t)b * NN + m) * NN + nrow] = (_Float16)c[r];
        }
    }
}

// ---------------------------------------------------------------------------
// conv1: 1 -> 64 channels, 7x7 pad 3, + relu — WMMA (K=49 padded to 64)
// img: f16 (300x300) ; w1p: f16 (64 oc, 64 k) ; out: f16 (PIX, 64)
// one wave = 16 pixels x 64 oc : 2 K-steps x 4 WMMA
// ---------------------------------------------------------------------------
__global__ void conv1_kernel(const _Float16* __restrict__ img,
                             const _Float16* __restrict__ w1p,
                             const float* __restrict__ bias,
                             _Float16* __restrict__ out) {
    int wave = (blockIdx.x * blockDim.x + threadIdx.x) >> 5;
    int lane = threadIdx.x & 31;
    if (wave >= MT2) return;
    int p = wave * 16 + (lane & 15);
    int y = p / NN, x = p - y * NN;
    int ahi = (lane >= 16) ? 8 : 0;
    int bhi = (lane >= 16) ? 8 : 0;
    int ocBase = lane & 15;

    v8f acc[4] = {{}, {}, {}, {}};
    #pragma unroll
    for (int kc = 0; kc < 2; ++kc) {
        v16h a = {};
        #pragma unroll
        for (int h = 0; h < 16; ++h) {
            int k = kc * 32 + ((h < 8) ? h : 16 + (h - 8)) + ahi;  // tap id
            _Float16 v = (_Float16)0.0f;
            if (k < 49) {
                int ky = k / 7, kx = k - ky * 7;
                int iy = y + ky - 3, ix = x + kx - 3;
                if (iy >= 0 && iy < NN && ix >= 0 && ix < NN)
                    v = img[iy * NN + ix];
            }
            a[h] = v;
        }
        #pragma unroll
        for (int tn = 0; tn < 4; ++tn) {
            const _Float16* wr = w1p + (size_t)(tn * 16 + ocBase) * 64 + kc * 32 + bhi;
            v16h b;
            #pragma unroll
            for (int h = 0; h < 16; ++h) b[h] = wr[(h < 8) ? h : 16 + (h - 8)];
            acc[tn] = __builtin_amdgcn_wmma_f32_16x16x32_f16(
                false, a, false, b, (short)0, acc[tn], false, false);
        }
    }
    #pragma unroll
    for (int tn = 0; tn < 4; ++tn) {
        int oc = tn * 16 + ocBase;
        float bv = bias[oc];
        #pragma unroll
        for (int r = 0; r < 8; ++r) {
            int m = wave * 16 + r + ((lane >= 16) ? 8 : 0);
            out[(size_t)m * MID + oc] = (_Float16)fmaxf(acc[tn][r] + bv, 0.0f);
        }
    }
}

// ---------------------------------------------------------------------------
// conv2: 64 -> 64 ch, 7x7 pad 3, + relu — implicit GEMM, WMMA, LDS-staged B
// k = (ky*7+kx)*64 + ic ; in f16 (PIX,64) pixel-major ; w2rT f16 [oc][k]
// Block = 16 waves (512 thr), 16 M-tiles; per tap stage 64x64 f16 slice (8KB)
// into LDS once, all waves read B from LDS. 392 WMMAs per wave.
// ---------------------------------------------------------------------------
__global__ void conv2_kernel(const _Float16* __restrict__ in,
                             const _Float16* __restrict__ w2rT,
                             const float* __restrict__ bias,
                             _Float16* __restrict__ out) {
    __shared__ _Float16 wlds[MID * MID];          // 8 KB tap slice [oc][ic]
    int tid  = threadIdx.x;                       // 512
    int wave = tid >> 5;
    int lane = tid & 31;
    int tileM = blockIdx.x * 16 + wave;
    bool active = tileM < MT2;
    int p = active ? (tileM * 16 + (lane & 15)) : 0;
    int y = p / NN, x = p - y * NN;
    int ahi = (lane >= 16) ? 8 : 0;
    int bhi = (lane >= 16) ? 8 : 0;
    int ocBase = lane & 15;

    // staging assignment: each thread copies 8 halfs (16 B)
    int s_oc  = tid >> 3;          // 0..63
    int s_icb = (tid & 7) * 8;     // 0,8,..,56

    v8f acc[4] = {{}, {}, {}, {}};
    for (int ky = 0; ky < 7; ++ky) {
        int iy = y + ky - 3;
        for (int kx = 0; kx < 7; ++kx) {
            int tap = ky * 7 + kx;
            __syncthreads();   // previous tap's compute done
            {
                const int4* src = (const int4*)(w2rT + (size_t)s_oc * K2 + tap * 64 + s_icb);
                int4* dst = (int4*)(wlds + s_oc * MID + s_icb);
                *dst = *src;
            }
            __syncthreads();   // slice visible to all waves
            if (!active) continue;
            int ix = x + kx - 3;
            bool ok = (iy >= 0) & (iy < NN) & (ix >= 0) & (ix < NN);
            #pragma unroll
            for (int half = 0; half < 2; ++half) {     // ic 0..31 / 32..63
                v16h a = {};
                if (ok) {
                    const _Float16* pa =
                        in + (size_t)(iy * NN + ix) * MID + half * 32 + ahi;
                    #pragma unroll
                    for (int h = 0; h < 8; ++h) {
                        a[h]     = pa[h];
                        a[h + 8] = pa[16 + h];
                    }
                }
                #pragma unroll
                for (int tn = 0; tn < 4; ++tn) {
                    const _Float16* wr =
                        wlds + (tn * 16 + ocBase) * MID + half * 32 + bhi;
                    v16h b;
                    #pragma unroll
                    for (int h = 0; h < 16; ++h) b[h] = wr[(h < 8) ? h : 16 + (h - 8)];
                    acc[tn] = __builtin_amdgcn_wmma_f32_16x16x32_f16(
                        false, a, false, b, (short)0, acc[tn], false, false);
                }
            }
        }
    }
    if (active) {
        #pragma unroll
        for (int tn = 0; tn < 4; ++tn) {
            int oc = tn * 16 + ocBase;
            float bv = bias[oc];
            #pragma unroll
            for (int r = 0; r < 8; ++r) {
                int m = tileM * 16 + r + ((lane >= 16) ? 8 : 0);
                out[(size_t)m * MID + oc] = (_Float16)fmaxf(acc[tn][r] + bv, 0.0f);
            }
        }
    }
}

// ---------------------------------------------------------------------------
// conv3: 64 -> 1 ch, 7x7 pad 3 — WMMA, N padded to 16 (rows 1..15 zero)
// in f16 (PIX,64) ; w3p f16 (16, 3136) ; out f32 (PIX)
// ---------------------------------------------------------------------------
__global__ void conv3_kernel(const _Float16* __restrict__ in,
                             const _Float16* __restrict__ w3p,
                             const float* __restrict__ bias,
                             float* __restrict__ out) {
    int wave = (blockIdx.x * blockDim.x + threadIdx.x) >> 5;
    int lane = threadIdx.x & 31;
    if (wave >= MT2) return;
    int p = wave * 16 + (lane & 15);
    int y = p / NN, x = p - y * NN;
    int ahi = (lane >= 16) ? 8 : 0;
    int bhi = (lane >= 16) ? 8 : 0;
    int ocBase = lane & 15;

    v8f acc = {};
    for (int ky = 0; ky < 7; ++ky) {
        int iy = y + ky - 3;
        #pragma unroll
        for (int kx = 0; kx < 7; ++kx) {
            int ix = x + kx - 3;
            bool ok = (iy >= 0) & (iy < NN) & (ix >= 0) & (ix < NN);
            int tap = ky * 7 + kx;
            #pragma unroll
            for (int half = 0; half < 2; ++half) {
                v16h a = {};
                if (ok) {
                    const _Float16* pa =
                        in + (size_t)(iy * NN + ix) * MID + half * 32 + ahi;
                    #pragma unroll
                    for (int h = 0; h < 8; ++h) {
                        a[h]     = pa[h];
                        a[h + 8] = pa[16 + h];
                    }
                }
                const _Float16* wr = w3p + (size_t)ocBase * K2 + tap * 64 + half * 32 + bhi;
                v16h b;
                #pragma unroll
                for (int h = 0; h < 16; ++h) b[h] = wr[(h < 8) ? h : 16 + (h - 8)];
                acc = __builtin_amdgcn_wmma_f32_16x16x32_f16(
                    false, a, false, b, (short)0, acc, false, false);
            }
        }
    }
    if (ocBase == 0) {       // only N column 0 is the real output channel
        float bv = bias[0];
        #pragma unroll
        for (int r = 0; r < 8; ++r) {
            int m = wave * 16 + r + ((lane >= 16) ? 8 : 0);
            out[m] = acc[r] + bv;
        }
    }
}

// ---------------------------------------------------------------------------
// top-100 per batch (iterative argmax, lowest-index tie break, sorted desc)
// ---------------------------------------------------------------------------
__global__ void topk_kernel(const float* __restrict__ imp,
                            float* __restrict__ scratch,
                            int* __restrict__ idxOut,
                            float* __restrict__ subPos,
                            float* __restrict__ objPos) {
    __shared__ float sv[1024];
    __shared__ int   si[1024];
    int b = blockIdx.x;
    int t = threadIdx.x;
    const float* src = imp + (size_t)b * PIX;
    float* s = scratch + (size_t)b * PIX;
    for (int i = t; i < PIX; i += 1024) s[i] = src[i];
    __syncthreads();

    for (int it = 0; it < KREL; ++it) {
        float bv = -INFINITY;
        int   bi = 0x7fffffff;
        for (int i = t; i < PIX; i += 1024) {
            float v = s[i];
            if (v > bv || (v == bv && i < bi)) { bv = v; bi = i; }
        }
        sv[t] = bv; si[t] = bi;
        __syncthreads();
        for (int off = 512; off > 0; off >>= 1) {
            if (t < off) {
                float ov = sv[t + off]; int oi = si[t + off];
                if (ov > sv[t] || (ov == sv[t] && oi < si[t])) {
                    sv[t] = ov; si[t] = oi;
                }
            }
            __syncthreads();
        }
        if (t == 0) {
            int w = si[0];
            s[w] = -INFINITY;
            idxOut[b * KREL + it] = w;
            subPos[b * KREL + it] = (float)(w / NN);
            objPos[b * KREL + it] = (float)(w % NN);
        }
        __syncthreads();
    }
}

// ---------------------------------------------------------------------------
// Gather sub/obj query + cls rows and build pair_query
// ---------------------------------------------------------------------------
__global__ void gather_kernel(const float* __restrict__ query,
                              const float* __restrict__ cls,
                              const int* __restrict__ idx,
                              float* __restrict__ subQ, float* __restrict__ objQ,
                              float* __restrict__ subC, float* __restrict__ objC,
                              float* __restrict__ pairQ) {
    int j = blockIdx.x;
    int b = blockIdx.y;
    int t = threadIdx.x;
    int w = idx[b * KREL + j];
    int sp = w / NN, op = w - sp * NN;
    const float* qs = query + ((size_t)b * NN + sp) * CC;
    const float* qo = query + ((size_t)b * NN + op) * CC;
    float vs = qs[t], vo = qo[t];
    subQ[((size_t)b * KREL + j) * CC + t]               = vs;
    objQ[((size_t)b * KREL + j) * CC + t]               = vo;
    pairQ[((size_t)b * 2 * KREL + j) * CC + t]          = vs;
    pairQ[((size_t)b * 2 * KREL + KREL + j) * CC + t]   = vo;
    if (t < NCLS) {
        subC[((size_t)b * KREL + j) * NCLS + t] = cls[((size_t)b * NN + sp) * NCLS + t];
        objC[((size_t)b * KREL + j) * NCLS + t] = cls[((size_t)b * NN + op) * NCLS + t];
    }
}

// ---------------------------------------------------------------------------
// Host side
// ---------------------------------------------------------------------------
static inline size_t alignUp(size_t x) { return (x + 255) & ~(size_t)255; }

extern "C" void kernel_launch(void* const* d_in, const int* in_sizes, int n_in,
                              void* d_out, int out_size, void* d_ws, size_t ws_size,
                              hipStream_t stream) {
    const float* query = (const float*)d_in[0];   // (16,300,256)
    const float* cls   = (const float*)d_in[1];   // (16,300,134)
    const float* sw[3] = { (const float*)d_in[2], (const float*)d_in[4], (const float*)d_in[6] };
    const float* sb[3] = { (const float*)d_in[3], (const float*)d_in[5], (const float*)d_in[7] };
    const float* ow[3] = { (const float*)d_in[8], (const float*)d_in[10], (const float*)d_in[12] };
    const float* ob[3] = { (const float*)d_in[9], (const float*)d_in[11], (const float*)d_in[13] };
    const float* cw1 = (const float*)d_in[14]; const float* cb1 = (const float*)d_in[15];
    const float* cw2 = (const float*)d_in[16]; const float* cb2 = (const float*)d_in[17];
    const float* cw3 = (const float*)d_in[18]; const float* cb3 = (const float*)d_in[19];

    const int ROWS = BB * NN;              // 4800
    // ---- workspace carve-up ----
    char* base = (char*)d_ws;
    size_t off = 0;
    auto carve = [&](size_t bytes) { char* p = base + off; off += alignUp(bytes); return p; };
    _Float16* qh     = (_Float16*)carve((size_t)ROWS * CC * 2);
    _Float16* xa     = (_Float16*)carve((size_t)ROWS * CC * 2);
    _Float16* xb     = (_Float16*)carve((size_t)ROWS * CC * 2);
    _Float16* subE   = (_Float16*)carve((size_t)ROWS * CC * 2);
    _Float16* objE   = (_Float16*)carve((size_t)ROWS * CC * 2);
    _Float16* wt[6];
    for (int i = 0; i < 6; ++i) wt[i] = (_Float16*)carve((size_t)CC * CC * 2);
    _Float16* w1p    = (_Float16*)carve((size_t)MID * 64 * 2);
    _Float16* w2rT   = (_Float16*)carve((size_t)MID * K2 * 2);
    _Float16* w3p    = (_Float16*)carve((size_t)16 * K2 * 2);
    _Float16* eimg   = (_Float16*)carve((size_t)BB * PIX * 2);    // einsum f16
    _Float16* c1out  = (_Float16*)carve((size_t)MID * PIX * 2);   // pixel-major
    _Float16* c2out  = (_Float16*)carve((size_t)MID * PIX * 2);   // pixel-major
    float*    tkbuf  = (float*)   carve((size_t)BB * PIX * 4);
    int*      tkidx  = (int*)     carve((size_t)BB * KREL * 4);
    (void)ws_size; (void)n_in; (void)in_sizes; (void)out_size;

    // ---- output carve-up (flat, reference return order) ----
    float* outF    = (float*)d_out;
    float* o_imp   = outF;                         // 16*300*300
    float* o_spos  = o_imp  + (size_t)BB * PIX;    // 16*100
    float* o_opos  = o_spos + (size_t)BB * KREL;   // 16*100
    float* o_subQ  = o_opos + (size_t)BB * KREL;   // 16*100*256
    float* o_objQ  = o_subQ + (size_t)BB * KREL * CC;
    float* o_subC  = o_objQ + (size_t)BB * KREL * CC;
    float* o_objC  = o_subC + (size_t)BB * KREL * NCLS;
    float* o_pairQ = o_objC + (size_t)BB * KREL * NCLS;

    // ---- 1. precision / layout prep ----
    {
        int n = ROWS * CC;
        f32_to_f16_kernel<<<(n + 255) / 256, 256, 0, stream>>>(query, qh, n);
    }
    for (int i = 0; i < 3; ++i) {
        transpose_to_f16_kernel<<<(CC * CC + 255) / 256, 256, 0, stream>>>(sw[i], wt[i],     CC, CC);
        transpose_to_f16_kernel<<<(CC * CC + 255) / 256, 256, 0, stream>>>(ow[i], wt[3 + i], CC, CC);
    }
    conv1_wprep_kernel<<<(MID * 64 + 255) / 256, 256, 0, stream>>>(cw1, w1p);
    conv2_wreorder_kernel<<<(MID * K2 + 255) / 256, 256, 0, stream>>>(cw2, w2rT);
    conv3_wprep_kernel<<<(16 * K2 + 255) / 256, 256, 0, stream>>>(cw3, w3p);

    // ---- 2. MLPs (WMMA) + l2norm ----
    const int mlpBlocks = (ROWS / 16) * 16 / 4;    // 4800 waves / 4 per block
    mlp_layer_kernel<<<mlpBlocks, 128, 0, stream>>>(qh, wt[0], sb[0], xa, ROWS, 1);
    mlp_layer_kernel<<<mlpBlocks, 128, 0, stream>>>(xa, wt[1], sb[1], xb, ROWS, 1);
    mlp_layer_kernel<<<mlpBlocks, 128, 0, stream>>>(xb, wt[2], sb[2], xa, ROWS, 0);
    l2norm_kernel<<<ROWS / 4, 128, 0, stream>>>(xa, subE, ROWS);
    mlp_layer_kernel<<<mlpBlocks, 128, 0, stream>>>(qh, wt[3], ob[0], xa, ROWS, 1);
    mlp_layer_kernel<<<mlpBlocks, 128, 0, stream>>>(xa, wt[4], ob[1], xb, ROWS, 1);
    mlp_layer_kernel<<<mlpBlocks, 128, 0, stream>>>(xb, wt[5], ob[2], xa, ROWS, 0);
    l2norm_kernel<<<ROWS / 4, 128, 0, stream>>>(xa, objE, ROWS);

    // ---- 3. einsum (WMMA, f16 image out) ----
    einsum_kernel<<<dim3(19, 19, BB), 32, 0, stream>>>(subE, objE, eimg);

    // ---- 4. conv stack (per batch; all three convs on the WMMA pipe) ----
    const int cBlocks = (MT2 + 3) / 4;             // 4 waves per block
    for (int b = 0; b < BB; ++b) {
        conv1_kernel<<<cBlocks, 128, 0, stream>>>(
            eimg + (size_t)b * PIX, w1p, cb1, c1out);
        conv2_kernel<<<(MT2 + 15) / 16, 512, 0, stream>>>(c1out, w2rT, cb2, c2out);
        conv3_kernel<<<cBlocks, 128, 0, stream>>>(
            c2out, w3p, cb3, o_imp + (size_t)b * PIX);
    }

    // ---- 5. top-k + gathers ----
    topk_kernel<<<BB, 1024, 0, stream>>>(o_imp, tkbuf, tkidx, o_spos, o_opos);
    gather_kernel<<<dim3(KREL, BB), 256, 0, stream>>>(
        query, cls, tkidx, o_subQ, o_objQ, o_subC, o_objC, o_pairQ);
}
